// HierarchicalGroupStageMoE_41841571398183
// MI455X (gfx1250) — matrix operand order repaired and checked by reading, not compiled, not run on hardware
//
#include <hip/hip_runtime.h>
#include <math.h>

// ---------------------------------------------------------------------------
// HierarchicalGroupStageMoE for MI455X (gfx1250, wave32, WMMA bf16)
//
// Sparse MoE dispatch (4 active experts/token instead of 32) + bf16 WMMA for
// every 128-wide GEMM. GEMM waves are register-blocked 2Mx4N (32x64 output,
// 8 WMMAs per 12 b128 loads) so they are WMMA-throughput bound, not L2 bound.
// All launches use fixed grids (graph-capture safe) and output accumulation
// is deterministic (per-pair partial buffer + reduce). Workspace: ~110 MB.
// ---------------------------------------------------------------------------

typedef __attribute__((ext_vector_type(16))) __bf16 v16bf;
typedef __attribute__((ext_vector_type(8)))  __bf16 v8bf;
typedef __attribute__((ext_vector_type(8)))  float  v8f;

#define N_TOK 4096
#define DM    512
#define F_DIM 128
#define G_GRP 8
#define FG    16
#define DH    128
#define DE    256
#define ES    4
#define E_EXP 32
#define NPAIR (N_TOK * 4)

// ---------------------------------------------------------------- helpers --
__device__ __forceinline__ float geluf(float x) {
    // jax.nn.gelu default (approximate=True, tanh form)
    float x3 = x * x * x;
    float t  = tanhf(0.7978845608028654f * (x + 0.044715f * x3));
    return 0.5f * x * (1.0f + t);
}

// A-fragment (16x32 bf16, one row per lane pair-half). Lane L: row = L%16,
// half = L/16. Element j: K = j + 8*half (j<8) / 16 + (j-8) + 8*half (j>=8).
__device__ __forceinline__ v16bf load_a_frag(const __bf16* __restrict__ p, int half) {
    v8bf lo = *(const v8bf*)(p + half * 8);
    v8bf hi = *(const v8bf*)(p + 16 + half * 8);
    v16bf r;
#pragma unroll
    for (int j = 0; j < 8; ++j) { r[j] = lo[j]; r[j + 8] = hi[j]; }
    return r;
}

// B-fragment: pre-packed so each lane reads 16 contiguous bf16 (32 bytes).
__device__ __forceinline__ v16bf load_b_frag(const __bf16* __restrict__ p) {
    v8bf lo = *(const v8bf*)(p);
    v8bf hi = *(const v8bf*)(p + 8);
    v16bf r;
#pragma unroll
    for (int j = 0; j < 8; ++j) { r[j] = lo[j]; r[j + 8] = hi[j]; }
    return r;
}

__device__ __forceinline__ v8f wmma_bf16(v16bf a, v16bf b, v8f c) {
    return __builtin_amdgcn_wmma_f32_16x16x32_bf16(false, a, false, b,
                                                   (short)0, c, false, false);
}

// --------------------------------------------------------------- LN fused --
// One block per token: LN over DM=512 with two (gain,bias) applications.
__global__ __launch_bounds__(128) void ln_hidden_kernel(
    const float* __restrict__ hidden,
    const float* __restrict__ pg, const float* __restrict__ pb,
    const float* __restrict__ rg, const float* __restrict__ rb,
    __bf16* __restrict__ hn, __bf16* __restrict__ hln)
{
    __shared__ float ssum[128], ssq[128];
    __shared__ float s_mean, s_inv;
    int n = blockIdx.x, t = threadIdx.x;
    float x[4]; float s = 0.f, q = 0.f;
#pragma unroll
    for (int i = 0; i < 4; ++i) {
        x[i] = hidden[(long)n * DM + t + 128 * i];
        s += x[i]; q += x[i] * x[i];
    }
    ssum[t] = s; ssq[t] = q; __syncthreads();
    for (int o = 64; o > 0; o >>= 1) {
        if (t < o) { ssum[t] += ssum[t + o]; ssq[t] += ssq[t + o]; }
        __syncthreads();
    }
    if (t == 0) {
        float mean = ssum[0] * (1.0f / DM);
        float var  = ssq[0] * (1.0f / DM) - mean * mean;
        s_mean = mean; s_inv = rsqrtf(var + 1e-5f);
    }
    __syncthreads();
    float mean = s_mean, inv = s_inv;
#pragma unroll
    for (int i = 0; i < 4; ++i) {
        int c = t + 128 * i;
        float v = (x[i] - mean) * inv;
        hn [(long)n * DM + c] = (__bf16)(v * pg[c] + pb[c]);
        hln[(long)n * DM + c] = (__bf16)(v * rg[c] + rb[c]);
    }
}

// One block per token: full LN over F=128 + per-group LN over FG=16
// (group output zero-padded to 32 columns so WMMA K-tiles are uniform).
__global__ __launch_bounds__(128) void ln_feat_kernel(
    const float* __restrict__ feat,
    const float* __restrict__ sg, const float* __restrict__ sb,
    const float* __restrict__ gg, const float* __restrict__ gb,
    __bf16* __restrict__ fln, __bf16* __restrict__ glnpad)
{
    __shared__ float ssum[128], ssq[128];
    __shared__ float s_mean, s_inv;
    int n = blockIdx.x, t = threadIdx.x;
    float x = feat[(long)n * F_DIM + t];
    ssum[t] = x; ssq[t] = x * x; __syncthreads();
    for (int o = 64; o > 0; o >>= 1) {
        if (t < o) { ssum[t] += ssum[t + o]; ssq[t] += ssq[t + o]; }
        __syncthreads();
    }
    if (t == 0) {
        float mean = ssum[0] * (1.0f / F_DIM);
        float var  = ssq[0] * (1.0f / F_DIM) - mean * mean;
        s_mean = mean; s_inv = rsqrtf(var + 1e-5f);
    }
    __syncthreads();
    // per-group LN via 16-lane shuffles (wave32: two groups per wave)
    float gs = x, gq = x * x;
    for (int m = 1; m < 16; m <<= 1) {
        gs += __shfl_xor(gs, m, 16);
        gq += __shfl_xor(gq, m, 16);
    }
    float gm = gs * (1.0f / FG);
    float ginv = rsqrtf(gq * (1.0f / FG) - gm * gm + 1e-5f);
    int g = t >> 4, f = t & 15;
    fln[(long)n * F_DIM + t] = (__bf16)((x - s_mean) * s_inv * sg[t] + sb[t]);
    long gbase = (long)n * (G_GRP * 32) + g * 32;
    glnpad[gbase + f]      = (__bf16)((x - gm) * ginv * gg[t] + gb[t]);
    glnpad[gbase + 16 + f] = (__bf16)0.0f;   // K padding 16 -> 32
}

// ----------------------------------------------------------- weight pack --
// Pack W[b][K][Ncol] fp32 -> bf16 B-fragment layout:
//   idx = (((b*Ntiles + tn)*Ktiles + tk)*32 + lane)*16 + j
//   value = W[b][tk*32 + 8*(lane/16) + (j%8) + 16*(j/8)][tn*16 + lane%16]   (0 if k>=K)
__global__ void pack_b_kernel(const float* __restrict__ W, __bf16* __restrict__ out,
                              int K, int Kpad, int Ncol, long total)
{
    int Kt = Kpad >> 5, Nt = Ncol >> 4;
    for (long i = (long)blockIdx.x * blockDim.x + threadIdx.x; i < total;
         i += (long)gridDim.x * blockDim.x) {
        int  j    = (int)(i & 15);
        int  lane = (int)((i >> 4) & 31);
        long rest = i >> 9;
        int  tk   = (int)(rest % Kt); rest /= Kt;
        int  tn   = (int)(rest % Nt);
        long b    = rest / Nt;
        int  k    = tk * 32 + 8 * (lane >> 4) + (j & 7) + 16 * (j >> 3);
        int  col  = tn * 16 + (lane & 15);
        float v = (k < K) ? W[b * (long)K * Ncol + (long)k * Ncol + col] : 0.0f;
        out[i] = (__bf16)v;
    }
}

// -------------------------------------------------------- generic WMMA GEMM
// One wave per 32x64 C-block (2 M-tiles x 4 N-tiles -> 8 accumulators).
// Per K-step: 2 A-frag + 4 B-frag loads feed 8 WMMAs.
// grid = (Mrows/32, Ncols/64, batch).
__global__ __launch_bounds__(32) void gemm_bf16_kernel(
    const __bf16* __restrict__ A, int aRowStride, int aBatchOff,
    const __bf16* __restrict__ Bp,
    const float* __restrict__ bias, int biasStride,
    __bf16* __restrict__ C, int cRowStride, int cBatchOff,
    int Ktiles, int Ntiles, int doGelu)
{
    const int lane = threadIdx.x, m = lane & 15, half = lane >> 4;
    const int mblk = blockIdx.x, nblk = blockIdx.y, z = blockIdx.z;
    const int row0 = mblk * 32 + m;
    const __bf16* arow0 = A + (long)z * aBatchOff + (long)row0 * aRowStride;
    const __bf16* arow1 = arow0 + (long)16 * aRowStride;
    const __bf16* bb = Bp + ((((long)z * Ntiles + nblk * 4) * Ktiles) * 32 + lane) * 16;
    v8f acc[8];
#pragma unroll
    for (int i = 0; i < 8; ++i)
#pragma unroll
        for (int r = 0; r < 8; ++r) acc[i][r] = 0.0f;
    for (int tk = 0; tk < Ktiles; ++tk) {
        v16bf a0 = load_a_frag(arow0 + tk * 32, half);
        v16bf a1 = load_a_frag(arow1 + tk * 32, half);
#pragma unroll
        for (int nt = 0; nt < 4; ++nt) {
            v16bf b = load_b_frag(bb + ((long)nt * Ktiles + tk) * 512);
            acc[nt]     = wmma_bf16(a0, b, acc[nt]);
            acc[4 + nt] = wmma_bf16(a1, b, acc[4 + nt]);
        }
    }
#pragma unroll
    for (int nt = 0; nt < 4; ++nt) {
        const int col = (nblk * 4 + nt) * 16 + m;
        const float bv = bias[z * biasStride + col];
#pragma unroll
        for (int mi = 0; mi < 2; ++mi)
#pragma unroll
            for (int r = 0; r < 8; ++r) {
                int orow = mblk * 32 + mi * 16 + r + 8 * half;
                float v = acc[mi * 4 + nt][r] + bv;
                if (doGelu) v = geluf(v);
                C[(long)z * cBatchOff + (long)orow * cRowStride + col] = (__bf16)v;
            }
    }
}

// --------------------------------------------- router_in fused WMMA GEMM --
// router_in[n,g] = concat(h_enc[n], s_enc[n], g_enc[n,g], h_enc[n]*g_enc[n,g])
// A-fragments synthesized on the fly (no 32 MB concat buffer), amortized
// over 4 N-tiles and 2 M-tiles per wave.
// mode 0: rows = n*8+g over N*G (pgs_w1, single B/bias)
// mode 1: rows = n, batch z = g    (ir_w1, per-group B/bias)  Ktiles = 16.
__global__ __launch_bounds__(32) void gemm_routerin_kernel(
    const __bf16* __restrict__ h_enc, const __bf16* __restrict__ s_enc,
    const __bf16* __restrict__ g_enc,
    const __bf16* __restrict__ Bp, const float* __restrict__ bias,
    __bf16* __restrict__ out, int mode)
{
    const int lane = threadIdx.x, m = lane & 15, half = lane >> 4;
    const int mblk = blockIdx.x, nblk = blockIdx.y, z = blockIdx.z;
    int n0, g0, n1, g1;
    if (mode == 0) {
        int rid0 = mblk * 32 + m;      n0 = rid0 >> 3; g0 = rid0 & 7;
        int rid1 = mblk * 32 + 16 + m; n1 = rid1 >> 3; g1 = rid1 & 7;
    } else {
        n0 = mblk * 32 + m; g0 = blockIdx.z;
        n1 = n0 + 16;       g1 = blockIdx.z;
    }
    const __bf16* ph0 = h_enc + (long)n0 * DH;
    const __bf16* ps0 = s_enc + (long)n0 * DH;
    const __bf16* pg0 = g_enc + ((long)n0 * G_GRP + g0) * DH;
    const __bf16* ph1 = h_enc + (long)n1 * DH;
    const __bf16* ps1 = s_enc + (long)n1 * DH;
    const __bf16* pg1 = g_enc + ((long)n1 * G_GRP + g1) * DH;
    const long bsel0 = (mode == 0) ? (long)(nblk * 4) : ((long)z * 8 + nblk * 4);
    const __bf16* bb = Bp + ((bsel0 * 16) * 32 + lane) * 16;   // Ktiles = 16
    v8f acc[8];
#pragma unroll
    for (int i = 0; i < 8; ++i)
#pragma unroll
        for (int r = 0; r < 8; ++r) acc[i][r] = 0.0f;
    for (int tk = 0; tk < 16; ++tk) {
        int seg = tk >> 2, koff = (tk & 3) * 32;
        v16bf a0, a1;
        if      (seg == 0) { a0 = load_a_frag(ph0 + koff, half); a1 = load_a_frag(ph1 + koff, half); }
        else if (seg == 1) { a0 = load_a_frag(ps0 + koff, half); a1 = load_a_frag(ps1 + koff, half); }
        else if (seg == 2) { a0 = load_a_frag(pg0 + koff, half); a1 = load_a_frag(pg1 + koff, half); }
        else {
            v16bf h0 = load_a_frag(ph0 + koff, half);
            v16bf q0 = load_a_frag(pg0 + koff, half);
            v16bf h1 = load_a_frag(ph1 + koff, half);
            v16bf q1 = load_a_frag(pg1 + koff, half);
#pragma unroll
            for (int j = 0; j < 16; ++j) {
                a0[j] = (__bf16)((float)h0[j] * (float)q0[j]);
                a1[j] = (__bf16)((float)h1[j] * (float)q1[j]);
            }
        }
#pragma unroll
        for (int nt = 0; nt < 4; ++nt) {
            v16bf b = load_b_frag(bb + ((long)nt * 16 + tk) * 512);
            acc[nt]     = wmma_bf16(a0, b, acc[nt]);
            acc[4 + nt] = wmma_bf16(a1, b, acc[4 + nt]);
        }
    }
#pragma unroll
    for (int nt = 0; nt < 4; ++nt) {
        const int col = (nblk * 4 + nt) * 16 + m;
        const float bv = bias[(mode == 0 ? 0 : z * DH) + col];
#pragma unroll
        for (int mi = 0; mi < 2; ++mi)
#pragma unroll
            for (int r = 0; r < 8; ++r) {
                int rid_out = mblk * 32 + mi * 16 + r + 8 * half;
                long orow = (mode == 0) ? (long)rid_out : ((long)rid_out * G_GRP + z);
                out[orow * DH + col] = (__bf16)geluf(acc[mi * 4 + nt][r] + bv);
            }
    }
}

// -------------------------------------------------- scoring + topk routing
__global__ void topk_kernel(
    const __bf16* __restrict__ pg_h, const float* __restrict__ pgs_w2,
    const float* __restrict__ pgs_b2,
    const __bf16* __restrict__ ir_h, const float* __restrict__ ir_w2,
    const float* __restrict__ ir_b2,
    int* __restrict__ topk_idx, float* __restrict__ topk_w)
{
    int n = blockIdx.x * blockDim.x + threadIdx.x;
    if (n >= N_TOK) return;
    float gl[G_GRP];
    float il[G_GRP][ES];
    for (int g = 0; g < G_GRP; ++g) {
        const __bf16* p = pg_h + ((long)n * G_GRP + g) * DH;
        float a = pgs_b2[0];
        for (int i = 0; i < DH; ++i) a += (float)p[i] * pgs_w2[i];
        gl[g] = a;                       // TEMP == 1.0
        const __bf16* q = ir_h + ((long)n * G_GRP + g) * DH;
        float b0 = ir_b2[g * ES + 0], b1 = ir_b2[g * ES + 1];
        float b2 = ir_b2[g * ES + 2], b3 = ir_b2[g * ES + 3];
        for (int i = 0; i < DH; ++i) {
            float h = (float)q[i];
            const float* w = ir_w2 + ((long)g * DH + i) * ES;
            b0 += h * w[0]; b1 += h * w[1]; b2 += h * w[2]; b3 += h * w[3];
        }
        il[g][0] = b0; il[g][1] = b1; il[g][2] = b2; il[g][3] = b3;
    }
    // top-2 groups (ties -> lower index, like jax.lax.top_k)
    int g0 = 0;
    for (int g = 1; g < G_GRP; ++g) if (gl[g] > gl[g0]) g0 = g;
    int g1 = (g0 == 0) ? 1 : 0;
    for (int g = 0; g < G_GRP; ++g) if (g != g0 && gl[g] > gl[g1]) g1 = g;
    float e1 = __expf(gl[g1] - gl[g0]);
    float gw[2] = { 1.0f / (1.0f + e1), e1 / (1.0f + e1) };
    int gs[2] = { g0, g1 };
    for (int s = 0; s < 2; ++s) {
        int g = gs[s];
        int a = 0;
        for (int e = 1; e < ES; ++e) if (il[g][e] > il[g][a]) a = e;
        int b = (a == 0) ? 1 : 0;
        for (int e = 0; e < ES; ++e) if (e != a && il[g][e] > il[g][b]) b = e;
        float ee = __expf(il[g][b] - il[g][a]);
        float w0 = 1.0f / (1.0f + ee), w1 = ee / (1.0f + ee);
        topk_idx[n * 4 + s * 2 + 0] = g * ES + a;
        topk_w  [n * 4 + s * 2 + 0] = gw[s] * w0;
        topk_idx[n * 4 + s * 2 + 1] = g * ES + b;
        topk_w  [n * 4 + s * 2 + 1] = gw[s] * w1;
    }
}

__global__ void zero_counts_kernel(int* counts) {
    if (threadIdx.x < E_EXP) counts[threadIdx.x] = 0;
}

__global__ void dispatch_kernel(const int* __restrict__ topk_idx,
                                int* __restrict__ counts, int* __restrict__ list)
{
    int p = blockIdx.x * blockDim.x + threadIdx.x;
    if (p >= NPAIR) return;
    int e = topk_idx[p];
    int pos = atomicAdd(&counts[e], 1);
    list[(long)e * N_TOK + pos] = p;    // pair id = token*4 + slot
}

// ---------------------------------------- sparse expert GEMMs (gathered) --
// eh[pid] = gelu(hn[token(pid)] @ ex_w1[e] + ex_b1[e]); Ktiles=16, Ntiles=16
// 2Mx4N register blocking; worst-case grid with count-based early exit.
__global__ __launch_bounds__(32) void expert1_kernel(
    const __bf16* __restrict__ hn, const __bf16* __restrict__ w1p,
    const float* __restrict__ b1, const int* __restrict__ counts,
    const int* __restrict__ list, __bf16* __restrict__ eh)
{
    const int e = blockIdx.z;
    const int cnt = counts[e];
    const int base = blockIdx.x * 32;
    if (base >= cnt) return;
    const int lane = threadIdx.x, m = lane & 15, half = lane >> 4;
    int iA0 = base + m;      if (iA0 >= cnt) iA0 = cnt - 1;
    int iA1 = base + 16 + m; if (iA1 >= cnt) iA1 = cnt - 1;
    const __bf16* arow0 = hn + (long)(list[(long)e * N_TOK + iA0] >> 2) * DM;
    const __bf16* arow1 = hn + (long)(list[(long)e * N_TOK + iA1] >> 2) * DM;
    const __bf16* bb = w1p + (long)e * (DM * DE)
                     + ((((long)blockIdx.y * 4) * 16) * 32 + lane) * 16;
    v8f acc[8];
#pragma unroll
    for (int i = 0; i < 8; ++i)
#pragma unroll
        for (int r = 0; r < 8; ++r) acc[i][r] = 0.0f;
    for (int tk = 0; tk < 16; ++tk) {
        v16bf a0 = load_a_frag(arow0 + tk * 32, half);
        v16bf a1 = load_a_frag(arow1 + tk * 32, half);
#pragma unroll
        for (int nt = 0; nt < 4; ++nt) {
            v16bf b = load_b_frag(bb + ((long)nt * 16 + tk) * 512);
            acc[nt]     = wmma_bf16(a0, b, acc[nt]);
            acc[4 + nt] = wmma_bf16(a1, b, acc[4 + nt]);
        }
    }
#pragma unroll
    for (int nt = 0; nt < 4; ++nt) {
        const int col = (blockIdx.y * 4 + nt) * 16 + m;
        const float bv = b1[e * DE + col];
#pragma unroll
        for (int mi = 0; mi < 2; ++mi)
#pragma unroll
            for (int r = 0; r < 8; ++r) {
                int ri = base + mi * 16 + r + 8 * half;
                if (ri < cnt) {
                    int pid = list[(long)e * N_TOK + ri];
                    eh[(long)pid * DE + col] = (__bf16)geluf(acc[mi * 4 + nt][r] + bv);
                }
            }
    }
}

// out_partial[pid] = topk_w[pid] * (eh[pid] @ ex_w2[e] + ex_b2[e]); Kt=8,Nt=32
__global__ __launch_bounds__(32) void expert2_kernel(
    const __bf16* __restrict__ eh, const __bf16* __restrict__ w2p,
    const float* __restrict__ b2, const int* __restrict__ counts,
    const int* __restrict__ list, const float* __restrict__ topk_w,
    float* __restrict__ part)
{
    const int e = blockIdx.z;
    const int cnt = counts[e];
    const int base = blockIdx.x * 32;
    if (base >= cnt) return;
    const int lane = threadIdx.x, m = lane & 15, half = lane >> 4;
    int iA0 = base + m;      if (iA0 >= cnt) iA0 = cnt - 1;
    int iA1 = base + 16 + m; if (iA1 >= cnt) iA1 = cnt - 1;
    const __bf16* arow0 = eh + (long)list[(long)e * N_TOK + iA0] * DE;
    const __bf16* arow1 = eh + (long)list[(long)e * N_TOK + iA1] * DE;
    const __bf16* bb = w2p + (long)e * (DE * DM)
                     + ((((long)blockIdx.y * 4) * 8) * 32 + lane) * 16;
    v8f acc[8];
#pragma unroll
    for (int i = 0; i < 8; ++i)
#pragma unroll
        for (int r = 0; r < 8; ++r) acc[i][r] = 0.0f;
    for (int tk = 0; tk < 8; ++tk) {
        v16bf a0 = load_a_frag(arow0 + tk * 32, half);
        v16bf a1 = load_a_frag(arow1 + tk * 32, half);
#pragma unroll
        for (int nt = 0; nt < 4; ++nt) {
            v16bf b = load_b_frag(bb + ((long)nt * 8 + tk) * 512);
            acc[nt]     = wmma_bf16(a0, b, acc[nt]);
            acc[4 + nt] = wmma_bf16(a1, b, acc[4 + nt]);
        }
    }
#pragma unroll
    for (int nt = 0; nt < 4; ++nt) {
        const int col = (blockIdx.y * 4 + nt) * 16 + m;
        const float bv = b2[e * DM + col];
#pragma unroll
        for (int mi = 0; mi < 2; ++mi)
#pragma unroll
            for (int r = 0; r < 8; ++r) {
                int ri = base + mi * 16 + r + 8 * half;
                if (ri < cnt) {
                    int pid = list[(long)e * N_TOK + ri];
                    part[(long)pid * DM + col] = topk_w[pid] * (acc[mi * 4 + nt][r] + bv);
                }
            }
    }
}

__global__ void reduce_kernel(const float* __restrict__ part, float* __restrict__ out)
{
    long i = (long)blockIdx.x * blockDim.x + threadIdx.x;   // over N*DM
    long n = i >> 9; int d = (int)(i & 511);
    const float* p = part + (n * 4) * DM + d;
    out[i] = p[0] + p[DM] + p[2 * DM] + p[3 * DM];
}

// ------------------------------------------------------------------ launch
extern "C" void kernel_launch(void* const* d_in, const int* in_sizes, int n_in,
                              void* d_out, int out_size, void* d_ws, size_t ws_size,
                              hipStream_t stream)
{
    (void)in_sizes; (void)n_in; (void)out_size; (void)ws_size;
    const float* hidden = (const float*)d_in[0];
    const float* feat   = (const float*)d_in[1];
    const float* pre_g  = (const float*)d_in[2];
    const float* pre_b  = (const float*)d_in[3];
    const float* rh_g   = (const float*)d_in[4];
    const float* rh_b   = (const float*)d_in[5];
    const float* rh_w1  = (const float*)d_in[6];
    const float* rh_b1  = (const float*)d_in[7];
    const float* rh_w2  = (const float*)d_in[8];
    const float* rh_b2  = (const float*)d_in[9];
    const float* sf_g   = (const float*)d_in[10];
    const float* sf_b   = (const float*)d_in[11];
    const float* sf_w1  = (const float*)d_in[12];
    const float* sf_b1  = (const float*)d_in[13];
    const float* sf_w2  = (const float*)d_in[14];
    const float* sf_b2  = (const float*)d_in[15];
    const float* gf_g   = (const float*)d_in[16];
    const float* gf_b   = (const float*)d_in[17];
    const float* gf_w1  = (const float*)d_in[18];
    const float* gf_b1  = (const float*)d_in[19];
    const float* gf_w2  = (const float*)d_in[20];
    const float* gf_b2  = (const float*)d_in[21];
    const float* pgs_w1 = (const float*)d_in[22];
    const float* pgs_b1 = (const float*)d_in[23];
    const float* pgs_w2 = (const float*)d_in[24];
    const float* pgs_b2 = (const float*)d_in[25];
    const float* ir_w1  = (const float*)d_in[26];
    const float* ir_b1  = (const float*)d_in[27];
    const float* ir_w2  = (const float*)d_in[28];
    const float* ir_b2  = (const float*)d_in[29];
    const float* ex_w1  = (const float*)d_in[30];
    const float* ex_b1  = (const float*)d_in[31];
    const float* ex_w2  = (const float*)d_in[32];
    const float* ex_b2  = (const float*)d_in[33];
    float* out = (float*)d_out;

    size_t off = 0;
    auto alloc = [&](size_t bytes) -> void* {
        void* p = (char*)d_ws + off;
        off += (bytes + 255) & ~(size_t)255;
        return p;
    };
    // activations (bf16)
    __bf16* hn     = (__bf16*)alloc((size_t)N_TOK * DM * 2);
    __bf16* hln    = (__bf16*)alloc((size_t)N_TOK * DM * 2);
    __bf16* fln    = (__bf16*)alloc((size_t)N_TOK * F_DIM * 2);
    __bf16* glnpad = (__bf16*)alloc((size_t)N_TOK * G_GRP * 32 * 2);
    __bf16* h1     = (__bf16*)alloc((size_t)N_TOK * DH * 2);
    __bf16* h_enc  = (__bf16*)alloc((size_t)N_TOK * DH * 2);
    __bf16* s1     = (__bf16*)alloc((size_t)N_TOK * DH * 2);
    __bf16* s_enc  = (__bf16*)alloc((size_t)N_TOK * DH * 2);
    __bf16* g1     = (__bf16*)alloc((size_t)N_TOK * G_GRP * DH * 2);
    __bf16* g_enc  = (__bf16*)alloc((size_t)N_TOK * G_GRP * DH * 2);
    __bf16* pg_h   = (__bf16*)alloc((size_t)N_TOK * G_GRP * DH * 2);
    __bf16* ir_h   = (__bf16*)alloc((size_t)N_TOK * G_GRP * DH * 2);
    __bf16* eh     = (__bf16*)alloc((size_t)NPAIR * DE * 2);
    // routing state
    int*   topk_idx = (int*)alloc((size_t)NPAIR * 4);
    float* topk_w   = (float*)alloc((size_t)NPAIR * 4);
    int*   counts   = (int*)alloc((size_t)E_EXP * 4);
    int*   list     = (int*)alloc((size_t)E_EXP * N_TOK * 4);
    float* part     = (float*)alloc((size_t)NPAIR * DM * 4);
    // packed weights (bf16, WMMA B-fragment layout)
    __bf16* rh_w1p  = (__bf16*)alloc((size_t)DM * DH * 2);
    __bf16* rh_w2p  = (__bf16*)alloc((size_t)DH * DH * 2);
    __bf16* sf_w1p  = (__bf16*)alloc((size_t)F_DIM * DH * 2);
    __bf16* sf_w2p  = (__bf16*)alloc((size_t)DH * DH * 2);
    __bf16* gf_w1p  = (__bf16*)alloc((size_t)G_GRP * 32 * DH * 2);
    __bf16* gf_w2p  = (__bf16*)alloc((size_t)G_GRP * DH * DH * 2);
    __bf16* pgs_w1p = (__bf16*)alloc((size_t)(4 * DH) * DH * 2);
    __bf16* ir_w1p  = (__bf16*)alloc((size_t)G_GRP * (4 * DH) * DH * 2);
    __bf16* ex_w1p  = (__bf16*)alloc((size_t)E_EXP * DM * DE * 2);
    __bf16* ex_w2p  = (__bf16*)alloc((size_t)E_EXP * DE * DM * 2);

    auto pack = [&](const float* W, __bf16* dst, int K, int Kpad, int Ncol, int batch) {
        long total = (long)batch * Kpad * Ncol;
        int blocks = (int)((total + 255) / 256);
        pack_b_kernel<<<dim3(blocks), dim3(256), 0, stream>>>(W, dst, K, Kpad, Ncol, total);
    };
    pack(rh_w1,  rh_w1p,  DM,     DM,     DH, 1);
    pack(rh_w2,  rh_w2p,  DH,     DH,     DH, 1);
    pack(sf_w1,  sf_w1p,  F_DIM,  F_DIM,  DH, 1);
    pack(sf_w2,  sf_w2p,  DH,     DH,     DH, 1);
    pack(gf_w1,  gf_w1p,  FG,     32,     DH, G_GRP);    // K padded 16 -> 32
    pack(gf_w2,  gf_w2p,  DH,     DH,     DH, G_GRP);
    pack(pgs_w1, pgs_w1p, 4 * DH, 4 * DH, DH, 1);
    pack(ir_w1,  ir_w1p,  4 * DH, 4 * DH, DH, G_GRP);
    pack(ex_w1,  ex_w1p,  DM,     DM,     DE, E_EXP);
    pack(ex_w2,  ex_w2p,  DE,     DE,     DM, E_EXP);

    // LayerNorms (fused gain/bias pairs)
    ln_hidden_kernel<<<dim3(N_TOK), dim3(128), 0, stream>>>(
        hidden, pre_g, pre_b, rh_g, rh_b, hn, hln);
    ln_feat_kernel<<<dim3(N_TOK), dim3(128), 0, stream>>>(
        feat, sf_g, sf_b, gf_g, gf_b, fln, glnpad);

    auto gemm = [&](const __bf16* A, int aRS, int aBO, const __bf16* Bp,
                    const float* bias, int biasStride, __bf16* C, int cRS, int cBO,
                    int Mrows, int Ncols, int batch, int Ktiles, int gelu) {
        gemm_bf16_kernel<<<dim3(Mrows / 32, Ncols / 64, batch), dim3(32), 0, stream>>>(
            A, aRS, aBO, Bp, bias, biasStride, C, cRS, cBO, Ktiles, Ncols / 16, gelu);
    };
    // router_hidden_encoder
    gemm(hln, DM, 0, rh_w1p, rh_b1, 0, h1,    DH, 0, N_TOK, DH, 1, DM / 32, 1);
    gemm(h1,  DH, 0, rh_w2p, rh_b2, 0, h_enc, DH, 0, N_TOK, DH, 1, DH / 32, 0);
    // stage_feature_encoder
    gemm(fln, F_DIM, 0, sf_w1p, sf_b1, 0, s1,    DH, 0, N_TOK, DH, 1, F_DIM / 32, 1);
    gemm(s1,  DH,    0, sf_w2p, sf_b2, 0, s_enc, DH, 0, N_TOK, DH, 1, DH / 32,   0);
    // group_feature_encoders (batched over G)
    gemm(glnpad, G_GRP * 32, 32, gf_w1p, gf_b1, DH, g1,    G_GRP * DH, DH,
         N_TOK, DH, G_GRP, 1, 1);
    gemm(g1, G_GRP * DH, DH, gf_w2p, gf_b2, DH, g_enc, G_GRP * DH, DH,
         N_TOK, DH, G_GRP, DH / 32, 0);
    // router_in fused GEMMs (pgs_w1 over N*G rows; ir_w1 batched per group)
    gemm_routerin_kernel<<<dim3(N_TOK * G_GRP / 32, DH / 64, 1), dim3(32), 0, stream>>>(
        h_enc, s_enc, g_enc, pgs_w1p, pgs_b1, pg_h, 0);
    gemm_routerin_kernel<<<dim3(N_TOK / 32, DH / 64, G_GRP), dim3(32), 0, stream>>>(
        h_enc, s_enc, g_enc, ir_w1p, ir_b1, ir_h, 1);
    // scoring + top-2/top-2 scatter-softmax routing
    topk_kernel<<<dim3(N_TOK / 256), dim3(256), 0, stream>>>(
        pg_h, pgs_w2, pgs_b2, ir_h, ir_w2, ir_b2, topk_idx, topk_w);
    // sparse dispatch
    zero_counts_kernel<<<dim3(1), dim3(32), 0, stream>>>(counts);
    dispatch_kernel<<<dim3(NPAIR / 256), dim3(256), 0, stream>>>(topk_idx, counts, list);
    // gathered expert MLPs (worst-case grid, early-exit on counts)
    expert1_kernel<<<dim3(N_TOK / 32, DE / 64, E_EXP), dim3(32), 0, stream>>>(
        hn, ex_w1p, ex_b1, counts, list, eh);
    expert2_kernel<<<dim3(N_TOK / 32, DM / 64, E_EXP), dim3(32), 0, stream>>>(
        eh, ex_w2p, ex_b2, counts, list, topk_w, part);
    // deterministic 4-way combine
    reduce_kernel<<<dim3(N_TOK * DM / 256), dim3(256), 0, stream>>>(part, out);
}
